// Attention_9715216023920
// MI455X (gfx1250) — matrix operand order repaired and checked by reading, not compiled, not run on hardware
//
#include <hip/hip_runtime.h>

typedef __bf16 bf16_t;
typedef __attribute__((ext_vector_type(16))) __bf16 v16bf;
typedef __attribute__((ext_vector_type(8)))  __bf16 v8bf;
typedef __attribute__((ext_vector_type(8)))  float  v8f;

#define HD    12
#define NSEQ  576
#define CDIM  768
#define DHEAD 64
#define BATCH 64
#define SROW  580   // padded f32 row stride for S in LDS (580%64==4 -> conflict-free)

// ---------------- WMMA fragment helpers (CDNA5 wave32 layouts) ----------------

// A-matrix 16x32 bf16, row-major source with leading dim `ld` (elements).
// lanes 0-15: row=lane, K = 0..7 then 16..23 ; lanes 16-31: row=lane-16, K = 8..15 then 24..31.
__device__ __forceinline__ v16bf load_a_frag(const bf16_t* __restrict__ base, int ld, int lane) {
  const int row = lane & 15;
  const int kb  = (lane >> 4) << 3;          // 0 or 8
  const bf16_t* p = base + (size_t)row * ld + kb;
  v8bf lo = *(const v8bf*)(p);
  v8bf hi = *(const v8bf*)(p + 16);
  v16bf f;
#pragma unroll
  for (int i = 0; i < 8; ++i) { f[i] = lo[i]; f[i + 8] = hi[i]; }
  return f;
}

// B-matrix 32x16 bf16 supplied as its transpose (row-major [N][K], leading dim `ld`).
// lanes 0-15: col=lane, K = 0..7 then 16..23 ; lanes 16-31: col=lane-16, K = 8..15 then 24..31.
__device__ __forceinline__ v16bf load_b_frag(const bf16_t* __restrict__ baset, int ld, int lane) {
  const int col = lane & 15;
  const int kb  = (lane >> 4) << 3;          // 0 or 8
  const bf16_t* p = baset + (size_t)col * ld + kb;
  v8bf lo = *(const v8bf*)(p);
  v8bf hi = *(const v8bf*)(p + 16);
  v16bf f;
#pragma unroll
  for (int i = 0; i < 8; ++i) { f[i] = lo[i]; f[i + 8] = hi[i]; }
  return f;
}

__device__ __forceinline__ v8f wmma_bf16(v16bf a, v16bf b, v8f c) {
  return __builtin_amdgcn_wmma_f32_16x16x32_bf16(false, a, false, b, (short)0, c, false, false);
}

// ---------------- Phase 0: conversions + bias gather ----------------

__global__ void cvt_kernel(const float* __restrict__ in, bf16_t* __restrict__ out, int n) {
  int i = blockIdx.x * blockDim.x + threadIdx.x;
  if (i < n) out[i] = (bf16_t)in[i];
}

__global__ void bias_kernel(const float* __restrict__ rpb, const int* __restrict__ rel,
                            float* __restrict__ biasF) {
  int i = blockIdx.x * blockDim.x + threadIdx.x;   // over 576*576
  if (i >= NSEQ * NSEQ) return;
  int r = rel[i];
#pragma unroll
  for (int h = 0; h < HD; ++h)
    biasF[(size_t)h * NSEQ * NSEQ + i] = rpb[(size_t)r * HD + h];
}

// ---------------- Phase 1: QKV GEMM (bf16 WMMA), routes Q/K/Vt ----------------

__global__ __launch_bounds__(256) void qkv_kernel(
    const bf16_t* __restrict__ xb, const bf16_t* __restrict__ wb,
    const float* __restrict__ q_bias, const float* __restrict__ v_bias,
    bf16_t* __restrict__ Q, bf16_t* __restrict__ K, bf16_t* __restrict__ Vt) {
  const int lane = threadIdx.x & 31;
  const int wave = threadIdx.x >> 5;
  const int mrow = blockIdx.y * 128 + wave * 16;   // 288 blocks * 128 rows = 36864
  const int ncol = blockIdx.x * 64;                // 36 blocks * 64 cols = 2304
  const bf16_t* abase = xb + (size_t)mrow * CDIM;
  v8f c[4] = {};
  for (int k = 0; k < CDIM; k += 32) {
    v16bf a = load_a_frag(abase + k, CDIM, lane);
#pragma unroll
    for (int nt = 0; nt < 4; ++nt) {
      v16bf bfrag = load_b_frag(wb + (size_t)(ncol + nt * 16) * CDIM + k, CDIM, lane);
      c[nt] = wmma_bf16(a, bfrag, c[nt]);
    }
  }
  const float scale = 0.125f;                      // 64^-0.5
  const int rhi = (lane >> 4) << 3;
  const int cl  = lane & 15;
#pragma unroll
  for (int nt = 0; nt < 4; ++nt) {
    const int jf = ncol + nt * 16 + cl;
#pragma unroll
    for (int r = 0; r < 8; ++r) {
      const int m  = mrow + rhi + r;
      const int bb = m / NSEQ, nn = m % NSEQ;
      float val = c[nt][r];
      if (jf < CDIM) {                                           // q
        val = (val + q_bias[jf]) * scale;
        const int hh = jf >> 6, dd = jf & 63;
        Q[(((size_t)bb * HD + hh) * NSEQ + nn) * DHEAD + dd] = (bf16_t)val;
      } else if (jf < 2 * CDIM) {                                // k (zero bias)
        const int f = jf - CDIM, hh = f >> 6, dd = f & 63;
        K[(((size_t)bb * HD + hh) * NSEQ + nn) * DHEAD + dd] = (bf16_t)val;
      } else {                                                   // v -> transposed
        const int f = jf - 2 * CDIM, hh = f >> 6, dd = f & 63;
        val += v_bias[f];
        Vt[(((size_t)bb * HD + hh) * DHEAD + dd) * NSEQ + nn] = (bf16_t)val;
      }
    }
  }
}

// ---------------- Phase 2: attention (scores + softmax + PV), per-wave 16 rows ----------------

__global__ __launch_bounds__(128) void attn_kernel(
    const bf16_t* __restrict__ Q, const bf16_t* __restrict__ Km,
    const bf16_t* __restrict__ Vt, const float* __restrict__ biasF,
    bf16_t* __restrict__ attnb) {
  extern __shared__ float smem[];
  const int lane = threadIdx.x & 31;
  const int wave = threadIdx.x >> 5;
  const int b = blockIdx.z, h = blockIdx.y;
  const int q0 = blockIdx.x * 64 + wave * 16;

  float* S    = smem + (size_t)wave * 16 * SROW;
  float* rinv = smem + (size_t)4 * 16 * SROW + wave * 16;

  const bf16_t* Qb = Q  + (((size_t)b * HD + h) * NSEQ + q0) * DHEAD;
  const bf16_t* Kb = Km + ((size_t)b * HD + h) * NSEQ * DHEAD;
  const bf16_t* Vb = Vt + ((size_t)b * HD + h) * DHEAD * NSEQ;
  const float*  Bh = biasF + (size_t)h * NSEQ * NSEQ;

  // Q fragments (reused for all 36 key tiles)
  v16bf qa0 = load_a_frag(Qb,      DHEAD, lane);
  v16bf qa1 = load_a_frag(Qb + 32, DHEAD, lane);

  const int rhi = (lane >> 4) << 3;
  const int cl  = lane & 15;

  // S = Q K^T  (f32 accumulate), spill per-wave row strip to LDS
  for (int t = 0; t < 36; ++t) {
    v8f c = {};
    v16bf kb0 = load_b_frag(Kb + (size_t)(t * 16) * DHEAD,      DHEAD, lane);
    c = wmma_bf16(qa0, kb0, c);
    v16bf kb1 = load_b_frag(Kb + (size_t)(t * 16) * DHEAD + 32, DHEAD, lane);
    c = wmma_bf16(qa1, kb1, c);
#pragma unroll
    for (int r = 0; r < 8; ++r)
      S[(size_t)(r + rhi) * SROW + t * 16 + cl] = c[r];
  }
  __syncthreads();

  // bias add + numerically stable softmax, one row at a time, 32 lanes cooperating
  for (int r = 0; r < 16; ++r) {
    float* row = S + (size_t)r * SROW;
    const float* br = Bh + (size_t)(q0 + r) * NSEQ;
    float m = -3.0e38f;
    for (int i = lane; i < NSEQ; i += 32) {
      float v = row[i] + br[i];
      row[i] = v;
      m = fmaxf(m, v);
    }
#pragma unroll
    for (int off = 16; off > 0; off >>= 1) m = fmaxf(m, __shfl_xor(m, off, 32));
    float s = 0.f;
    for (int i = lane; i < NSEQ; i += 32) {
      float p = __expf(row[i] - m);
      row[i] = p;
      s += p;
    }
#pragma unroll
    for (int off = 16; off > 0; off >>= 1) s += __shfl_xor(s, off, 32);
    if (lane == 0) rinv[r] = 1.0f / s;
  }
  __syncthreads();

  // O = P V : convert P f32->bf16 on the fly into A-fragments
  v8f o[4] = {};
  const int arow = lane & 15;
  const int akb  = (lane >> 4) << 3;
  for (int kc = 0; kc < NSEQ; kc += 32) {
    const float* sp = S + (size_t)arow * SROW + kc + akb;
    v16bf pa;
#pragma unroll
    for (int j = 0; j < 8; ++j) { pa[j] = (bf16_t)sp[j]; pa[j + 8] = (bf16_t)sp[j + 16]; }
#pragma unroll
    for (int nt = 0; nt < 4; ++nt) {
      v16bf vb = load_b_frag(Vb + (size_t)(nt * 16) * NSEQ + kc, NSEQ, lane);
      o[nt] = wmma_bf16(pa, vb, o[nt]);
    }
  }
#pragma unroll
  for (int nt = 0; nt < 4; ++nt) {
#pragma unroll
    for (int r = 0; r < 8; ++r) {
      const int rl = r + rhi;
      const float v = o[nt][r] * rinv[rl];
      attnb[((size_t)b * NSEQ + q0 + rl) * CDIM + h * DHEAD + nt * 16 + cl] = (bf16_t)v;
    }
  }
}

// ---------------- Phase 3: output projection GEMM, f32 out ----------------

__global__ __launch_bounds__(256) void proj_kernel(
    const bf16_t* __restrict__ ab, const bf16_t* __restrict__ wb,
    const float* __restrict__ pb, float* __restrict__ out) {
  const int lane = threadIdx.x & 31;
  const int wave = threadIdx.x >> 5;
  const int mrow = blockIdx.y * 128 + wave * 16;
  const int ncol = blockIdx.x * 64;               // 12 blocks * 64 = 768
  const bf16_t* abase = ab + (size_t)mrow * CDIM;
  v8f c[4] = {};
  for (int k = 0; k < CDIM; k += 32) {
    v16bf a = load_a_frag(abase + k, CDIM, lane);
#pragma unroll
    for (int nt = 0; nt < 4; ++nt) {
      v16bf bfrag = load_b_frag(wb + (size_t)(ncol + nt * 16) * CDIM + k, CDIM, lane);
      c[nt] = wmma_bf16(a, bfrag, c[nt]);
    }
  }
  const int rhi = (lane >> 4) << 3;
  const int cl  = lane & 15;
#pragma unroll
  for (int nt = 0; nt < 4; ++nt) {
    const int jf = ncol + nt * 16 + cl;
    const float bias = pb[jf];
#pragma unroll
    for (int r = 0; r < 8; ++r) {
      const int m = mrow + rhi + r;
      out[(size_t)m * CDIM + jf] = c[nt][r] + bias;
    }
  }
}

// ---------------- host launcher ----------------

extern "C" void kernel_launch(void* const* d_in, const int* in_sizes, int n_in,
                              void* d_out, int out_size, void* d_ws, size_t ws_size,
                              hipStream_t stream) {
  (void)in_sizes; (void)n_in; (void)out_size; (void)ws_size;
  const float* x      = (const float*)d_in[0];
  const float* qkv_w  = (const float*)d_in[1];
  const float* q_bias = (const float*)d_in[2];
  const float* v_bias = (const float*)d_in[3];
  const float* rpb    = (const float*)d_in[4];
  const int*   rel    = (const int*)d_in[5];
  const float* proj_w = (const float*)d_in[6];
  const float* proj_b = (const float*)d_in[7];
  float* out = (float*)d_out;

  char* ws = (char*)d_ws;
  size_t off = 0;
  auto carve = [&](size_t bytes) -> void* {
    void* p = ws + off;
    off = (off + bytes + 255) & ~(size_t)255;
    return p;
  };
  const size_t M = (size_t)BATCH * NSEQ;                     // 36864
  bf16_t* xb    = (bf16_t*)carve(M * CDIM * 2);              // x in bf16
  bf16_t* wqkv  = (bf16_t*)carve((size_t)3 * CDIM * CDIM * 2);
  bf16_t* wproj = (bf16_t*)carve((size_t)CDIM * CDIM * 2);
  bf16_t* Qd    = (bf16_t*)carve(M * CDIM * 2);              // [B,H,N,d] scaled
  bf16_t* Kd    = (bf16_t*)carve(M * CDIM * 2);              // [B,H,N,d]
  bf16_t* Vtd   = (bf16_t*)carve(M * CDIM * 2);              // [B,H,d,N]
  bf16_t* attnb = (bf16_t*)carve(M * CDIM * 2);              // [B,N,C] bf16
  float*  biasF = (float*)carve((size_t)HD * NSEQ * NSEQ * 4);

  const int nx = (int)(M * CDIM);
  cvt_kernel<<<(nx + 255) / 256, 256, 0, stream>>>(x, xb, nx);
  const int nw = 3 * CDIM * CDIM;
  cvt_kernel<<<(nw + 255) / 256, 256, 0, stream>>>(qkv_w, wqkv, nw);
  const int np = CDIM * CDIM;
  cvt_kernel<<<(np + 255) / 256, 256, 0, stream>>>(proj_w, wproj, np);
  bias_kernel<<<(NSEQ * NSEQ + 255) / 256, 256, 0, stream>>>(rpb, rel, biasF);

  qkv_kernel<<<dim3(36, 288), 256, 0, stream>>>(xb, wqkv, q_bias, v_bias, Qd, Kd, Vtd);

  const size_t shmem = ((size_t)4 * 16 * SROW + 64) * sizeof(float);   // ~148.7 KB (<320 KB/WGP)
  attn_kernel<<<dim3(9, HD, BATCH), 128, shmem, stream>>>(Qd, Kd, Vtd, biasF, attnb);

  proj_kernel<<<dim3(12, 288), 256, 0, stream>>>(attnb, wproj, proj_b, out);
}
